// SlotAllocator_14602888806499
// MI455X (gfx1250) — matrix operand order repaired and checked by reading, not compiled, not run on hardware
//
#include <hip/hip_runtime.h>

// ---------------------------------------------------------------------------
// Problem constants (from reference)
// ---------------------------------------------------------------------------
#define B_    4
#define T_    1024
#define N_    64
#define DS_   1024
#define DR_   512
#define HID_  128
#define KSEL  32          // top-k = clamp(N, 4, 32) = 32

typedef __attribute__((ext_vector_type(16))) __bf16 v16bf;
typedef __attribute__((ext_vector_type(8)))  __bf16 v8bf;
typedef __attribute__((ext_vector_type(8)))  float  v8f;
typedef int v4i __attribute__((vector_size(16)));

#if defined(__HIP_DEVICE_COMPILE__) && __has_builtin(__builtin_amdgcn_global_load_async_to_lds_b128)
#define HAVE_ASYNC_LDS 1
typedef __attribute__((address_space(1))) v4i* as1_v4i_ptr;
typedef __attribute__((address_space(3))) v4i* as3_v4i_ptr;
#endif

__device__ inline v8f zero_v8f() {
    v8f z;
#pragma unroll
    for (int i = 0; i < 8; ++i) z[i] = 0.0f;
    return z;
}

__device__ inline v16bf cat8(v8bf lo, v8bf hi) {
    v16bf a;
#pragma unroll
    for (int i = 0; i < 8; ++i) { a[i] = lo[i]; a[i + 8] = hi[i]; }
    return a;
}

__device__ inline void wait_async_zero() {
#ifdef HAVE_ASYNC_LDS
#if __has_builtin(__builtin_amdgcn_s_wait_asynccnt)
    __builtin_amdgcn_s_wait_asynccnt(0);
#else
    asm volatile("s_wait_asynccnt 0x0" ::: "memory");
#endif
#endif
}

// ---------------------------------------------------------------------------
// Kernel 1a: chunk sums for the prefix-mean scan.
//   s: (B,T,DS) fp32. T split into NCH chunks of TCH.
//   csum[b][ch][d] = sum over the chunk. grid = B*NCH*(DS/256), block 256.
// ---------------------------------------------------------------------------
#define NCH 8
#define TCH (T_ / NCH)          // 128

__global__ __launch_bounds__(256) void chunk_sum_kernel(const float* __restrict__ s,
                                                        float* __restrict__ csum) {
    const int dg  = DS_ / 256;                     // 4
    const int b   = blockIdx.x / (NCH * dg);
    const int ch  = (blockIdx.x / dg) % NCH;
    const int d   = (blockIdx.x % dg) * 256 + threadIdx.x;
    const float* sp = s + (size_t)b * T_ * DS_ + (size_t)ch * TCH * DS_ + d;
    float sum = 0.0f;
    for (int t = 0; t < TCH; ++t) sum += sp[(size_t)t * DS_];
    csum[((size_t)b * NCH + ch) * DS_ + d] = sum;
}

// ---------------------------------------------------------------------------
// Kernel 1b: scan within each chunk starting from the exclusive chunk prefix;
// emit context = running_sum / (t+1) as bf16.
// ---------------------------------------------------------------------------
__global__ __launch_bounds__(256) void prefix_mean_kernel(const float* __restrict__ s,
                                                          const float* __restrict__ csum,
                                                          __bf16* __restrict__ ctxbf) {
    const int dg  = DS_ / 256;
    const int b   = blockIdx.x / (NCH * dg);
    const int ch  = (blockIdx.x / dg) % NCH;
    const int d   = (blockIdx.x % dg) * 256 + threadIdx.x;
    float base = 0.0f;
    for (int c = 0; c < ch; ++c) base += csum[((size_t)b * NCH + c) * DS_ + d];
    const float* sp = s + (size_t)b * T_ * DS_ + (size_t)ch * TCH * DS_ + d;
    __bf16* cp = ctxbf + (size_t)b * T_ * DS_ + (size_t)ch * TCH * DS_ + d;
    float sum = base;
    for (int t = 0; t < TCH; ++t) {
        sum += sp[(size_t)t * DS_];
        cp[(size_t)t * DS_] = (__bf16)(sum / (float)(ch * TCH + t + 1));
    }
}

// ---------------------------------------------------------------------------
// Kernel 2: pack torch-layout weight W (O rows, K cols; y = x @ W^T) into
// per-WMMA-B-fragment bf16 layout (B 32x16 bf16):
//   fragment f = kStep*(O/16) + nTile; lane: n = nTile*16+(lane&15),
//   kHalf = lane>>4; element e -> k = kStep*32 + kHalf*8 + (e<8 ? e : e+8).
// ---------------------------------------------------------------------------
__global__ __launch_bounds__(256) void pack_weight_kernel(const float* __restrict__ W,
                                                          __bf16* __restrict__ outp,
                                                          int O, int K) {
    const int idx = blockIdx.x * blockDim.x + threadIdx.x;
    const int total = (K / 32) * (O / 16) * 32;
    if (idx >= total) return;
    const int lane = idx & 31;
    const int frag = idx >> 5;
    const int nTiles = O / 16;
    const int ks    = frag / nTiles;
    const int nTile = frag % nTiles;
    const int o     = nTile * 16 + (lane & 15);
    const int kBase = ks * 32 + (lane >> 4) * 8;
    const float* src = W + (size_t)o * K + kBase;
    __bf16* dst = outp + (size_t)idx * 16;
#pragma unroll
    for (int e = 0; e < 16; ++e) dst[e] = (__bf16)src[e < 8 ? e : e + 8];
}

// ---------------------------------------------------------------------------
// Kernel 3: ctx GEMM: ctxOut(BT x 512) = ctxbf(BT x 1024) x W_ctx^T (packed)
// ---------------------------------------------------------------------------
__global__ __launch_bounds__(256) void ctx_gemm_kernel(const __bf16* __restrict__ Abf,
                                                       const __bf16* __restrict__ Wp,
                                                       float* __restrict__ C) {
    const int rowBase = blockIdx.x * 64;
    const int wave = threadIdx.x >> 5;
    const int lane = threadIdx.x & 31;
    const int mTile = wave & 3;
    const int nHalf = wave >> 2;
    const int half  = lane >> 4;
    const int m = rowBase + mTile * 16 + (lane & 15);
    const __bf16* aRow = Abf + (size_t)m * DS_ + 8 * half;

    for (int pass = 0; pass < 2; ++pass) {
        v8f acc[8];
#pragma unroll
        for (int i = 0; i < 8; ++i) acc[i] = zero_v8f();

        for (int ks = 0; ks < DS_ / 32; ++ks) {
            v8bf lo = *(const v8bf*)(aRow + ks * 32);
            v8bf hi = *(const v8bf*)(aRow + ks * 32 + 16);
            v16bf a = cat8(lo, hi);
#pragma unroll
            for (int nt = 0; nt < 8; ++nt) {
                const int nTile = nHalf * 16 + pass * 8 + nt;
                v16bf b = *(const v16bf*)(Wp + ((size_t)(ks * (DR_ / 16) + nTile) * 32 + lane) * 16);
                acc[nt] = __builtin_amdgcn_wmma_f32_16x16x32_bf16(
                    false, a, false, b, (short)0, acc[nt], false, false);
            }
        }
#pragma unroll
        for (int nt = 0; nt < 8; ++nt) {
            const int nTile = nHalf * 16 + pass * 8 + nt;
            const int n = nTile * 16 + (lane & 15);
#pragma unroll
            for (int v = 0; v < 8; ++v) {
                const int mm = rowBase + mTile * 16 + v + 8 * half;
                C[(size_t)mm * DR_ + n] = acc[nt][v];
            }
        }
    }
}

// ---------------------------------------------------------------------------
// Kernel 4: fused per-(b,t) mega kernel.
//   GEMM1: slot_feat = r[bt](64x512) x W_slot^T + ctx[bt]  (WMMA bf16)
//          B fragments double-buffered in LDS via async global->LDS copies
//          (one 32KB stage per k-step shared by all 8 waves).
//   act = tanh(slot_feat) -> bf16 in LDS (64 x 520)
//   GEMM2: h = relu(act x W1^T + b1)  (k-outer, B frag loaded once per k-step)
//   scores = h.W2 + b2 via shfl_xor reduce; top-k + softmax straight-through.
// ---------------------------------------------------------------------------
#define TLD 520                          // Tbf row stride (bf16): 1040B = 65*16B
#define BSTG (32 * 32 * 16)              // bf16 elems per k-step stage = 16384 (32KB)

__global__ __launch_bounds__(256) void slot_mega_kernel(
    const float* __restrict__ r, const float* __restrict__ ctx,
    const __bf16* __restrict__ WslotP, const __bf16* __restrict__ W1P,
    const float* __restrict__ b1, const float* __restrict__ W2,
    const float* __restrict__ b2, float* __restrict__ out) {

    extern __shared__ __align__(16) char smemRaw[];
    __bf16* Bbuf = (__bf16*)smemRaw;                               // 2 x 32KB
    __bf16 (*Tbf)[TLD] = (__bf16 (*)[TLD])(smemRaw + 2 * BSTG * sizeof(__bf16));
    __shared__ float ctxRow[DR_];
    __shared__ float scPart[8][N_];
    __shared__ float sc[N_];

    const int bt  = blockIdx.x;
    const int tid = threadIdx.x;
    const float* rBase = r + (size_t)bt * N_ * DR_;

    for (int i = tid; i < DR_; i += 256) ctxRow[i] = ctx[(size_t)bt * DR_ + i];

    const int wave  = tid >> 5;
    const int lane  = tid & 31;
    const int mTile = wave & 3;
    const int nHalf = wave >> 2;
    const int half  = lane >> 4;
    const float* aRow = rBase + (size_t)(mTile * 16 + (lane & 15)) * DR_ + 8 * half;

    // ---- stage helper: copy 32KB of packed W_slot fragments for k-step ----
    auto stage = [&](int ks, int buf) {
        const __bf16* src = WslotP + (size_t)ks * BSTG;
        __bf16* dst = Bbuf + (size_t)buf * BSTG;
#pragma unroll
        for (int j = 0; j < 8; ++j) {
            const int c = tid + j * 256;                 // 16B chunk index
#ifdef HAVE_ASYNC_LDS
            __builtin_amdgcn_global_load_async_to_lds_b128(
                (as1_v4i_ptr)(src + c * 8),
                (as3_v4i_ptr)(dst + c * 8), 0, 0);
#else
            *(v8bf*)(dst + c * 8) = *(const v8bf*)(src + c * 8);
#endif
        }
    };

    // ---- GEMM1 with double-buffered LDS B staging ----
    v8f acc[16];
#pragma unroll
    for (int i = 0; i < 16; ++i) acc[i] = zero_v8f();

    stage(0, 0);
    for (int ks = 0; ks < DR_ / 32; ++ks) {
        const int cur = ks & 1;
        wait_async_zero();
        __syncthreads();                       // stage(ks) visible; prev buf free
        if (ks + 1 < DR_ / 32) stage(ks + 1, cur ^ 1);

        // A fragment: fp32 -> bf16 in registers
        v8f lo = *(const v8f*)(aRow + ks * 32);
        v8f hi = *(const v8f*)(aRow + ks * 32 + 16);
        v16bf a;
#pragma unroll
        for (int i = 0; i < 8; ++i) { a[i] = (__bf16)lo[i]; a[i + 8] = (__bf16)hi[i]; }

        const __bf16* bbase = Bbuf + (size_t)cur * BSTG;
#pragma unroll
        for (int nt = 0; nt < 16; ++nt) {
            const int nTile = nHalf * 16 + nt;
            v16bf b = *(const v16bf*)(bbase + ((size_t)nTile * 32 + lane) * 16);
            acc[nt] = __builtin_amdgcn_wmma_f32_16x16x32_bf16(
                false, a, false, b, (short)0, acc[nt], false, false);
        }
    }

    // epilogue: +ctx, tanh, store bf16 activations to LDS
#pragma unroll
    for (int nt = 0; nt < 16; ++nt) {
        const int nTile = nHalf * 16 + nt;
        const int n = nTile * 16 + (lane & 15);
        const float c = ctxRow[n];
#pragma unroll
        for (int v = 0; v < 8; ++v) {
            const int mm = mTile * 16 + v + 8 * half;
            Tbf[mm][n] = (__bf16)tanhf(acc[nt][v] + c);
        }
    }
    __syncthreads();

    // ---- GEMM2: k-outer so each W1 fragment is loaded exactly once ----
    const int oTile = wave;                       // 8 waves x 16 = 128 hidden
    const int o     = oTile * 16 + (lane & 15);
    const float w2v = W2[o];
    const float b1v = b1[o];

    v8f acc2[4];
#pragma unroll
    for (int i = 0; i < 4; ++i) acc2[i] = zero_v8f();

    for (int ks = 0; ks < DR_ / 32; ++ks) {
        v16bf b = *(const v16bf*)(W1P + ((size_t)(ks * (HID_ / 16) + oTile) * 32 + lane) * 16);
#pragma unroll
        for (int mT = 0; mT < 4; ++mT) {
            const __bf16* p = &Tbf[mT * 16 + (lane & 15)][ks * 32 + 8 * half];
            v16bf a = cat8(*(const v8bf*)p, *(const v8bf*)(p + 16));
            acc2[mT] = __builtin_amdgcn_wmma_f32_16x16x32_bf16(
                false, a, false, b, (short)0, acc2[mT], false, false);
        }
    }

#pragma unroll
    for (int mT = 0; mT < 4; ++mT) {
        float p[8];
#pragma unroll
        for (int v = 0; v < 8; ++v) {
            float h = acc2[mT][v] + b1v;
            h = h > 0.0f ? h : 0.0f;
            p[v] = h * w2v;
        }
#pragma unroll
        for (int mk = 1; mk < 16; mk <<= 1) {
#pragma unroll
            for (int v = 0; v < 8; ++v) p[v] += __shfl_xor(p[v], mk, 32);
        }
        if ((lane & 15) == 0) {
#pragma unroll
            for (int v = 0; v < 8; ++v)
                scPart[wave][mT * 16 + 8 * half + v] = p[v];
        }
    }
    __syncthreads();

    if (tid < N_) {
        float ssum = b2[0];
#pragma unroll
        for (int w = 0; w < 8; ++w) ssum += scPart[w][tid];
        sc[tid] = ssum;
    }
    __syncthreads();

    // ---- top-k hard mask + softmax straight-through (forward value) ----
    if (tid < N_) {
        const float si = sc[tid];
        int cnt = 0;
        float mx = -__builtin_inff();
        for (int j = 0; j < N_; ++j) {
            const float v = sc[j];
            cnt += (v > si) || (v == si && j < tid);
            mx = fmaxf(mx, v);
        }
        float sumE = 0.0f;
        for (int j = 0; j < N_; ++j) sumE += expf(sc[j] - mx);
        float soft = expf(si - mx) / sumE * (float)KSEL;
        soft = fminf(fmaxf(soft, 0.0f), 1.0f);
        const float hard = (cnt < KSEL) ? 1.0f : 0.0f;
        out[(size_t)bt * N_ + tid] = (hard + soft) - soft;
    }
}

// ---------------------------------------------------------------------------
// Launch
// ---------------------------------------------------------------------------
extern "C" void kernel_launch(void* const* d_in, const int* in_sizes, int n_in,
                              void* d_out, int out_size, void* d_ws, size_t ws_size,
                              hipStream_t stream) {
    (void)in_sizes; (void)n_in; (void)out_size; (void)ws_size;
    const float* s     = (const float*)d_in[0];
    const float* r     = (const float*)d_in[1];
    const float* Wctx  = (const float*)d_in[2];
    const float* Wslot = (const float*)d_in[3];
    const float* W1    = (const float*)d_in[4];
    const float* b1    = (const float*)d_in[5];
    const float* W2    = (const float*)d_in[6];
    const float* b2    = (const float*)d_in[7];
    float* out = (float*)d_out;

    // Workspace layout (bytes), 256B-aligned regions:
    //   ctxbf  : (B*T, DS) bf16  @        0, size 8388608
    //   WslotP : 512x512  bf16   @  8388608, size  524288
    //   W1P    : 128x512  bf16   @  8912896, size  131072
    //   WctxP  : 512x1024 bf16   @  9043968, size 1048576
    //   ctxOut : (B*T, DR) fp32  @ 10092544, size 8388608
    //   csum   : (B,NCH,DS) fp32 @ 18481152, size  131072
    char* ws = (char*)d_ws;
    __bf16* ctxbf  = (__bf16*)(ws);
    __bf16* WslotP = (__bf16*)(ws + 8388608);
    __bf16* W1P    = (__bf16*)(ws + 8912896);
    __bf16* WctxP  = (__bf16*)(ws + 9043968);
    float*  ctxOut = (float*)(ws + 10092544);
    float*  csum   = (float*)(ws + 18481152);

    // 1) two-phase prefix mean -> bf16 context
    chunk_sum_kernel<<<B_ * NCH * (DS_ / 256), 256, 0, stream>>>(s, csum);
    prefix_mean_kernel<<<B_ * NCH * (DS_ / 256), 256, 0, stream>>>(s, csum, ctxbf);

    // 2) pack weights into WMMA B-fragment layout (bf16)
    pack_weight_kernel<<<32768 / 256, 256, 0, stream>>>(Wctx, WctxP, DR_, DS_);
    pack_weight_kernel<<<16384 / 256, 256, 0, stream>>>(Wslot, WslotP, DR_, DR_);
    pack_weight_kernel<<<4096 / 256, 256, 0, stream>>>(W1, W1P, HID_, DR_);

    // 3) ctx GEMM: (B*T x DS) x (DS x DR)
    ctx_gemm_kernel<<<(B_ * T_) / 64, 256, 0, stream>>>(ctxbf, WctxP, ctxOut);

    // 4) fused slot-feature / MLP / top-k mask kernel, one block per (b,t)
    const size_t dynLds = 2 * BSTG * sizeof(__bf16)            // 64 KB B staging
                        + (size_t)N_ * TLD * sizeof(__bf16);   // 66560 B activations
    slot_mega_kernel<<<B_ * T_, 256, dynLds, stream>>>(
        r, ctxOut, WslotP, W1P, b1, W2, b2, out);
}
